// CrossAttention_54606214202168
// MI455X (gfx1250) — matrix operand order repaired and checked by reading, not compiled, not run on hardware
//
#include <hip/hip_runtime.h>
#include <hip/hip_bf16.h>

typedef __attribute__((ext_vector_type(16))) __bf16 v16bf;
typedef __attribute__((ext_vector_type(8)))  __bf16 v8bf;
typedef __attribute__((ext_vector_type(8)))  float  v8f;

#define HEADS 16
#define BB    4
#define NN    1024
#define MM    1024
#define MK    1025      // m + 1 (null token at j=0)
#define JS    1088      // padded V row stride (17*64, 32B-multiple in bf16)
#define INNER 1024

// ---------------------------------------------------------------------------
// WMMA helpers (gfx1250 wave32, 16x16x32 bf16 -> f32)
// ---------------------------------------------------------------------------
__device__ inline v8f bf16_wmma(v16bf a, v16bf b, v8f c) {
    return __builtin_amdgcn_wmma_f32_16x16x32_bf16(
        /*neg_a=*/false, a, /*neg_b=*/false, b,
        /*c_mod=*/(short)0, c, /*reuse_a=*/false, /*reuse_b=*/false);
}

// A-matrix fragment (16 rows x 32 K, bf16) from LDS tile [row][k].
// k order used here (contiguous, same as B): lane L(<16) = row L, holds K {h*16+0..15}, h=lane/16.
__device__ inline v16bf ld_afrag(const __bf16* base, int stride, int k0) {
    int lane = threadIdx.x & 31;
    int half = lane >> 4;
    int m    = lane & 15;
    const __bf16* p = base + m * stride + k0;
    v8bf lo = *(const v8bf*)(p + half * 16);
    v8bf hi = *(const v8bf*)(p + half * 16 + 8);
    v16bf r;
#pragma unroll
    for (int i = 0; i < 8; ++i) { r[i] = lo[i]; r[i + 8] = hi[i]; }
    return r;
}

// B-matrix fragment (32 K x 16 cols) from LDS tile stored TRANSPOSED [n][k].
// k order used here (contiguous, same as A): lane = column (lane&15), K = (lane/16)*16 + e -> contiguous 32B.
__device__ inline v16bf ld_bfrag(const __bf16* bt, int stride, int k0) {
    int lane = threadIdx.x & 31;
    int half = lane >> 4;
    int n    = lane & 15;
    return *(const v16bf*)(bt + n * stride + k0 + half * 16);
}

// ---------------------------------------------------------------------------
// CDNA5 async global->LDS copy (ASYNCcnt-tracked, no VGPR round trip).
// dsaddr = LDS byte address (low 32 bits of flat shared pointer);
// mem    = SGPR base + 32-bit VGPR byte offset (GVS mode).
// ---------------------------------------------------------------------------
__device__ inline void async_load_b128(unsigned lds_addr, const void* sbase, unsigned voff) {
    asm volatile("global_load_async_to_lds_b128 %0, %1, %2"
                 :: "v"(lds_addr), "v"(voff), "s"(sbase)
                 : "memory");
}

// ---------------------------------------------------------------------------
// Generic 64x64-tile GEMM, C = A(4096 x K) * B(K x Ccols), bf16 WMMA, f32 acc.
// EP 0: q-proj  (tanh -> q_ws[b][h][n][d] bf16)
// EP 1: kv-proj (k half: tanh -> k_ws[b][h][m+1][d]; v half -> v_ws[b][h][d][JS])
// EP 2: out-proj (A already bf16; + bias -> f32 d_out)
// ---------------------------------------------------------------------------
template<int EP, bool ABF16>
__global__ __launch_bounds__(128)
void gemm_wmma_kernel(const void* __restrict__ Aop, const float* __restrict__ Bmat,
                      int K, int Ccols,
                      __bf16* __restrict__ ws0, __bf16* __restrict__ ws1,
                      const float* __restrict__ bias, float* __restrict__ outf) {
    __shared__ __align__(32) __bf16 As[64 * 32];   // [m][k]
    __shared__ __align__(32) __bf16 Bts[64 * 32];  // [n][k] (transposed)

    const int tid  = threadIdx.x;
    const int wave = tid >> 5;
    const int lane = tid & 31;
    const int half = lane >> 4;
    const int nl   = lane & 15;
    const int r0   = blockIdx.x * 64;
    const int c0   = blockIdx.y * 64;
    const int wm   = (wave >> 1) * 32;
    const int wn   = (wave & 1) * 32;

    v8f acc[2][2] = {};

    for (int k0 = 0; k0 < K; k0 += 32) {
        {   // stage A tile 64x32 (f32 -> bf16 unless already bf16)
            int kk = tid & 31, mb = tid >> 5;
#pragma unroll
            for (int i = 0; i < 16; ++i) {
                int m = mb + i * 4;
                float f;
                if (ABF16) f = (float)((const __bf16*)Aop)[(size_t)(r0 + m) * K + k0 + kk];
                else       f = ((const float*)Aop)[(size_t)(r0 + m) * K + k0 + kk];
                As[m * 32 + kk] = (__bf16)f;
            }
        }
        {   // stage B tile 32x64 transposed into [n][k]
            int n = tid & 63, kb = tid >> 6;
#pragma unroll
            for (int i = 0; i < 16; ++i) {
                int kk = kb + i * 2;
                Bts[n * 32 + kk] = (__bf16)Bmat[(size_t)(k0 + kk) * Ccols + c0 + n];
            }
        }
        __syncthreads();

        v16bf a0 = ld_afrag(As + wm * 32, 32, 0);
        v16bf a1 = ld_afrag(As + (wm + 16) * 32, 32, 0);
        v16bf b0 = ld_bfrag(Bts + wn * 32, 32, 0);
        v16bf b1 = ld_bfrag(Bts + (wn + 16) * 32, 32, 0);
        acc[0][0] = bf16_wmma(a0, b0, acc[0][0]);
        acc[0][1] = bf16_wmma(a0, b1, acc[0][1]);
        acc[1][0] = bf16_wmma(a1, b0, acc[1][0]);
        acc[1][1] = bf16_wmma(a1, b1, acc[1][1]);
        __syncthreads();
    }

#pragma unroll
    for (int i = 0; i < 2; ++i)
#pragma unroll
        for (int j = 0; j < 2; ++j)
#pragma unroll
            for (int v = 0; v < 8; ++v) {
                int row = r0 + wm + i * 16 + v + half * 8;
                int col = c0 + wn + j * 16 + nl;
                float val = acc[i][j][v];
                if (EP == 0) {
                    int b = row >> 10, n = row & 1023;
                    int h = col >> 6,  d = col & 63;
                    ws0[(((size_t)b * HEADS + h) * NN + n) * 64 + d] = (__bf16)tanhf(val);
                } else if (EP == 1) {
                    int b = row >> 10, m = row & 1023;
                    if (col < INNER) {
                        int h = col >> 6, d = col & 63;
                        ws0[(((size_t)b * HEADS + h) * MK + (m + 1)) * 64 + d] = (__bf16)tanhf(val);
                    } else {
                        int c2 = col - INNER;
                        int h = c2 >> 6, d = c2 & 63;
                        // V stored TRANSPOSED: [b,h][d][JS], column m+1
                        ws1[(((size_t)b * HEADS + h) * 64 + d) * JS + (m + 1)] = (__bf16)val;
                    }
                } else {
                    outf[(size_t)row * INNER + col] = val + bias[col];
                }
            }
}

// Fill null-token (j=0) rows of k_ws / v_ws and zero the V stride padding.
__global__ void init_null_kernel(const float* __restrict__ null_key,
                                 const float* __restrict__ null_value,
                                 __bf16* __restrict__ k_ws, __bf16* __restrict__ v_ws) {
    int t = blockIdx.x * blockDim.x + threadIdx.x;
    if (t >= BB * HEADS * 64) return;
    int d  = t & 63;
    int bh = t >> 6;
    k_ws[(size_t)bh * MK * 64 + d] = (__bf16)tanhf(null_key[d]);
    __bf16* vrow = v_ws + ((size_t)bh * 64 + d) * JS;
    vrow[0] = (__bf16)null_value[d];
    for (int j = MK; j < JS; ++j) vrow[j] = (__bf16)0.f;   // so P(=0)*V(pad)=0 exactly
}

// ---------------------------------------------------------------------------
// Flash-style attention: block = 64 queries x one (b,h); 4 waves x 16 rows.
// K/V tiles double-buffered through LDS via global_load_async_to_lds_b128.
// ---------------------------------------------------------------------------
__global__ __launch_bounds__(128)
void attn_kernel(const __bf16* __restrict__ q_ws, const __bf16* __restrict__ k_ws,
                 const __bf16* __restrict__ v_ws,
                 const unsigned char* __restrict__ qmask,
                 const unsigned char* __restrict__ cmask,
                 __bf16* __restrict__ obuf) {
    __shared__ __align__(32) __bf16 Qs[64 * 64];          // [q][d]
    __shared__ __align__(32) __bf16 Ks[2][64 * 64];       // [j][d]  (B^T layout for S)
    __shared__ __align__(32) __bf16 Vts[2][64 * 64];      // [d][j]  (B^T layout for PV)
    __shared__ __align__(32) __bf16 Ps[4][16 * 64];       // per-wave P tile [q][j]

    const int tid  = threadIdx.x;
    const int wave = tid >> 5;
    const int lane = tid & 31;
    const int half = lane >> 4;
    const int nl   = lane & 15;
    const int bh   = blockIdx.y;          // b*HEADS + h
    const int b    = bh >> 4;
    const int h    = bh & 15;
    const int n0   = blockIdx.x * 64;

    const __bf16* kbase = k_ws + (size_t)bh * MK * 64;    // [j][d]
    const __bf16* vbase = v_ws + (size_t)bh * 64 * JS;    // [d][JS]

    // ---- async-issue Q tile (8KB contiguous) ----
    {
        const __bf16* qbase = q_ws + ((size_t)bh * NN + n0) * 64;
#pragma unroll
        for (int i = 0; i < 4; ++i) {
            int idx = tid + i * 128;                       // 0..511 b128 segments
            async_load_b128((unsigned)(uintptr_t)(Qs + idx * 8), qbase,
                            (unsigned)(idx * 16));
        }
    }
    // ---- async-issue K/V tile 0 (8 ops per thread) ----
    auto issue_tile = [&](int buf, int j0) {
#pragma unroll
        for (int i = 0; i < 4; ++i) {                      // K: contiguous 8KB
            int idx = tid + i * 128;
            async_load_b128((unsigned)(uintptr_t)(&Ks[buf][idx * 8]),
                            kbase + (size_t)j0 * 64, (unsigned)(idx * 16));
        }
#pragma unroll
        for (int i = 0; i < 4; ++i) {                      // V: 64 rows x 128B, stride JS*2
            int idx = tid + i * 128;
            int d = idx >> 3, seg = idx & 7;
            async_load_b128((unsigned)(uintptr_t)(&Vts[buf][d * 64 + seg * 8]),
                            vbase + (size_t)j0, (unsigned)(d * (JS * 2) + seg * 16));
        }
    };
    issue_tile(0, 0);
    asm volatile("s_wait_asynccnt 0x8" ::: "memory");      // Q landed (async in-order)
    __syncthreads();

    v16bf qa[2];
    qa[0] = ld_afrag(Qs + wave * 16 * 64, 64, 0);
    qa[1] = ld_afrag(Qs + wave * 16 * 64, 64, 32);

    float qv[8];
#pragma unroll
    for (int v = 0; v < 8; ++v) {
        int n = n0 + wave * 16 + v + half * 8;
        qv[v] = (qmask[b * NN + n] != 0) ? 1.f : 0.f;
    }

    v8f o[4] = {};
    float mrow[8], lrow[8];
#pragma unroll
    for (int v = 0; v < 8; ++v) { mrow[v] = -3e30f; lrow[v] = 0.f; }

    for (int jt = 0; jt < 17; ++jt) {
        const int cur = jt & 1;
        const int j0  = jt * 64;
        if (jt < 16) {
            issue_tile(cur ^ 1, j0 + 64);                  // prefetch next tile
            asm volatile("s_wait_asynccnt 0x8" ::: "memory");  // current tile landed
        } else {
            asm volatile("s_wait_asynccnt 0x0" ::: "memory");
        }
        __syncthreads();

        // ---- S = Q K^T ----
        v8f s[4];
#pragma unroll
        for (int t = 0; t < 4; ++t) {
            v8f z = {};
            z = bf16_wmma(qa[0], ld_bfrag(&Ks[cur][t * 16 * 64], 64, 0),  z);
            z = bf16_wmma(qa[1], ld_bfrag(&Ks[cur][t * 16 * 64], 64, 32), z);
            s[t] = z;
        }

        // ---- scale + mask (key invalid/OOB -> -2e30; query invalid -> -1e30) ----
        // Tail-tile over-reads produce finite garbage which is overwritten here
        // by SELECTION, so it never reaches the softmax.
#pragma unroll
        for (int t = 0; t < 4; ++t) {
            int j = j0 + t * 16 + nl;
            bool kv = (j < MK) && (j == 0 || cmask[b * MM + j - 1] != 0);
#pragma unroll
            for (int v = 0; v < 8; ++v) {
                float sv = s[t][v] * 0.125f;               // 1/sqrt(64)
                sv = (qv[v] != 0.f) ? sv : -1e30f;
                s[t][v] = kv ? sv : -2e30f;
            }
        }

        // ---- online softmax (16-lane xor-shuffle row reductions) ----
        float alpha[8];
#pragma unroll
        for (int v = 0; v < 8; ++v) {
            float x = fmaxf(fmaxf(s[0][v], s[1][v]), fmaxf(s[2][v], s[3][v]));
#pragma unroll
            for (int off = 8; off > 0; off >>= 1)
                x = fmaxf(x, __shfl_xor(x, off, 32));
            float mnew = fmaxf(mrow[v], x);
            alpha[v] = __expf(mrow[v] - mnew);
            mrow[v]  = mnew;
        }
#pragma unroll
        for (int v = 0; v < 8; ++v) {
            float acc = 0.f;
#pragma unroll
            for (int t = 0; t < 4; ++t) {
                float p = __expf(s[t][v] - mrow[v]);
                s[t][v] = p;
                acc += p;
            }
#pragma unroll
            for (int off = 8; off > 0; off >>= 1)
                acc += __shfl_xor(acc, off, 32);
            lrow[v] = lrow[v] * alpha[v] + acc;
        }
#pragma unroll
        for (int u = 0; u < 4; ++u)
#pragma unroll
            for (int v = 0; v < 8; ++v) o[u][v] *= alpha[v];

        // ---- P (C layout) -> per-wave LDS -> A layout (same-wave DS is in-order) ----
        __bf16* pw = &Ps[wave][0];
#pragma unroll
        for (int t = 0; t < 4; ++t)
#pragma unroll
            for (int v = 0; v < 8; ++v)
                pw[(v + half * 8) * 64 + t * 16 + nl] = (__bf16)s[t][v];

        // ---- O += P * V ----
#pragma unroll
        for (int c = 0; c < 2; ++c) {
            v16bf pa = ld_afrag(pw, 64, c * 32);
#pragma unroll
            for (int u = 0; u < 4; ++u)
                o[u] = bf16_wmma(pa, ld_bfrag(&Vts[cur][u * 16 * 64], 64, c * 32), o[u]);
        }
        __syncthreads();   // all waves done with buffer `cur` before it is re-filled
    }

    // ---- normalize + write O as bf16 [b*N + n][h*64 + d] ----
#pragma unroll
    for (int u = 0; u < 4; ++u)
#pragma unroll
        for (int v = 0; v < 8; ++v) {
            int n = n0 + wave * 16 + v + half * 8;
            int d = u * 16 + nl;
            float val = o[u][v] / lrow[v];
            obuf[(size_t)(b * NN + n) * INNER + h * 64 + d] = (__bf16)val;
        }
}

// ---------------------------------------------------------------------------
// Host launch
// ---------------------------------------------------------------------------
extern "C" void kernel_launch(void* const* d_in, const int* in_sizes, int n_in,
                              void* d_out, int out_size, void* d_ws, size_t ws_size,
                              hipStream_t stream) {
    const float*         x       = (const float*)d_in[0];
    const float*         context = (const float*)d_in[1];
    const unsigned char* mask    = (const unsigned char*)d_in[2];
    const unsigned char* cmask   = (const unsigned char*)d_in[3];
    const float*         Wq      = (const float*)d_in[4];
    const float*         Wkv     = (const float*)d_in[5];
    const float*         Wout    = (const float*)d_in[6];
    const float*         bout    = (const float*)d_in[7];
    const float*         null_k  = (const float*)d_in[8];
    const float*         null_v  = (const float*)d_in[9];
    float*               out     = (float*)d_out;

    // workspace carve-up (bf16), 256B aligned; k_ws padded one extra tile so
    // the attention tail tile's async over-read stays inside the allocation.
    char* w = (char*)d_ws;
    size_t off = 0;
    __bf16* q_ws = (__bf16*)(w + off); off += (size_t)BB * HEADS * NN * 64 * 2;            off = (off + 255) & ~(size_t)255;
    __bf16* k_ws = (__bf16*)(w + off); off += (size_t)BB * HEADS * MK * 64 * 2 + 8192;     off = (off + 255) & ~(size_t)255;
    __bf16* v_ws = (__bf16*)(w + off); off += (size_t)BB * HEADS * 64 * JS * 2;            off = (off + 255) & ~(size_t)255;
    __bf16* obuf = (__bf16*)(w + off);

    // 1) null token rows + V pad zeroing
    init_null_kernel<<<(BB * HEADS * 64 + 255) / 256, 256, 0, stream>>>(null_k, null_v, k_ws, v_ws);

    // 2) q = tanh(x @ Wq)        : 4096 x 1024 x 1024
    gemm_wmma_kernel<0, false><<<dim3(64, 16), 128, 0, stream>>>(
        x, Wq, 1024, 1024, q_ws, nullptr, nullptr, nullptr);

    // 3) kv = context @ Wkv      : 4096 x 1024 x 2048 (k tanh'ed, v transposed)
    gemm_wmma_kernel<1, false><<<dim3(64, 32), 128, 0, stream>>>(
        context, Wkv, 1024, 2048, k_ws, v_ws, nullptr, nullptr);

    // 4) attention (async double-buffered K/V)
    attn_kernel<<<dim3(NN / 64, BB * HEADS), 128, 0, stream>>>(
        q_ws, k_ws, v_ws, mask, cmask, obuf);

    // 5) out = O @ Wout + bout   : 4096 x 1024 x 1024, f32 result
    gemm_wmma_kernel<2, true><<<dim3(64, 16), 128, 0, stream>>>(
        obuf, Wout, 1024, 1024, nullptr, nullptr, bout, out);
}
